// EquivariantAttention_27333171872093
// MI455X (gfx1250) — matrix operand order repaired and checked by reading, not compiled, not run on hardware
//
#include <hip/hip_runtime.h>
#include <cstdint>

typedef __attribute__((ext_vector_type(8)))  float  v8f;
typedef __attribute__((ext_vector_type(16))) __bf16 v16bf;

union V16BF { v16bf v; unsigned short u[16]; };

#define E_EDGES 160000
#define N_NODES 10000
#define WAVES_PER_BLOCK 2
#define EDGES_PER_WAVE 16
#define EDGES_PER_BLOCK (WAVES_PER_BLOCK * EDGES_PER_WAVE)
#define PREP_ELEMS 51200   // (4 + 96) ops * 512 bf16

__device__ __forceinline__ unsigned short f2bf(float x) {
  unsigned int u = __float_as_uint(x);
  u += 0x7FFFu + ((u >> 16) & 1u);          // round-to-nearest-even
  return (unsigned short)(u >> 16);
}

__device__ __forceinline__ void atomicMaxF(float* addr, float v) {
  // ordered-int trick: works with -inf init, mixed signs
  if (v >= 0.0f) atomicMax((int*)addr, __float_as_int(v));
  else           atomicMin((unsigned int*)addr, __float_as_uint(v));
}

// ---------------------------------------------------------------------------
// Pre-swizzle W1 (64x32) and W2 (768x64) into bf16 WMMA B-operand blocks.
// B layout (16-bit, 32x16): lane = N (0-15) | N (16-31 group), slot j holds
// K = (j<8?j:j+8) + 8*(lane/16). One 512-element block per (N-tile, K-half).
// ---------------------------------------------------------------------------
__global__ __launch_bounds__(256) void ea_prep(const float* __restrict__ W1,
                                               const float* __restrict__ W2,
                                               unsigned short* __restrict__ Bswz) {
  int tid = blockIdx.x * blockDim.x + threadIdx.x;
  if (tid >= PREP_ELEMS) return;
  int op   = tid >> 9;          // block index
  int lane = (tid >> 4) & 31;
  int j    = tid & 15;
  int kb   = ((j < 8) ? j : (j + 8)) + ((lane >> 4) << 3);   // 0..31
  int nl   = lane & 15;
  float val;
  if (op < 4) {                        // W1: 4 N-tiles, K=32
    int n = op * 16 + nl;              // 0..63
    val = W1[n * 32 + kb];
  } else {                             // W2: 48 N-tiles x 2 K-halves
    int op2 = op - 4;
    int nt2 = op2 >> 1, kh = op2 & 1;
    int n = nt2 * 16 + nl;             // 0..767
    val = W2[n * 64 + kb + kh * 32];
  }
  Bswz[tid] = f2bf(val);
}

__global__ __launch_bounds__(256) void ea_init(float* __restrict__ out,
                                               float* __restrict__ smax,
                                               float* __restrict__ denom) {
  int tid = blockIdx.x * blockDim.x + threadIdx.x;
  if (tid < N_NODES * 24) out[tid] = 0.0f;
  if (tid < N_NODES * 4) { smax[tid] = -__builtin_inff(); denom[tid] = 0.0f; }
}

// ---------------------------------------------------------------------------
// Main fused kernel: MLP (WMMA bf16) + equivariant conv + scores per edge.
// One wave32 handles 16 edges.
// ---------------------------------------------------------------------------
__global__ __launch_bounds__(64) void ea_main(
    const int* __restrict__ src, const int* __restrict__ dst,
    const float* __restrict__ basis, const float* __restrict__ edge_feats,
    const float* __restrict__ f, const float* __restrict__ b1,
    const float* __restrict__ b2, const unsigned short* __restrict__ Bswz,
    float* __restrict__ scoresWs, float* __restrict__ vWs,
    float* __restrict__ smax)
{
  __shared__ float s_h[WAVES_PER_BLOCK][16 * 64];     // h activations (f32)
  __shared__ float s_tmp[WAVES_PER_BLOCK][16 * 96];   // tmp[e, m(0..31), d(0..2)]
  __shared__ float s_rw[WAVES_PER_BLOCK][16 * 32];    // rw tile staging
  __shared__ float s_conv[WAVES_PER_BLOCK][16 * 72];  // conv[e, c(0..23), d]

  const int lane = threadIdx.x & 31;
  const int w    = threadIdx.x >> 5;
  const int lm   = lane & 15;
  const int lh   = lane >> 4;                         // 0 or 1
  const int eBase = (blockIdx.x * WAVES_PER_BLOCK + w) * EDGES_PER_WAVE;

  float* Sh = s_h[w];
  float* St = s_tmp[w];
  float* Sr = s_rw[w];
  float* Sc = s_conv[w];

  // ---- GEMM1: h = relu(edge_feats @ W1^T + b1), one WMMA K-step ----------
  // A layout (16-bit 16x32): row M = lane%16; slot j -> K = (j<8?j:j+8)+8*lh
  V16BF a1;
  {
    const float* efRow = edge_feats + (size_t)(eBase + lm) * 32;
#pragma unroll
    for (int j = 0; j < 16; ++j) {
      int k = ((j < 8) ? j : (j + 8)) + lh * 8;
      a1.u[j] = f2bf(efRow[k]);
    }
  }
#pragma unroll
  for (int nt = 0; nt < 4; ++nt) {
    v16bf bmat = *(const v16bf*)(Bswz + nt * 512 + lane * 16);
    float bias = b1[nt * 16 + lm];
    v8f acc;
#pragma unroll
    for (int j = 0; j < 8; ++j) acc[j] = bias;
    acc = __builtin_amdgcn_wmma_f32_16x16x32_bf16(false, a1.v, false, bmat,
                                                  (short)0, acc, false, false);
#pragma unroll
    for (int j = 0; j < 8; ++j)   // C layout: row = j + 8*lh, col = nt*16+lm
      Sh[(j + 8 * lh) * 64 + nt * 16 + lm] = fmaxf(acc[j], 0.0f);
  }

  // ---- tmp[e,m,d] = sum_d1 f[src[e],m1,d1] * basis[e,d1,k], m=2*m1+k/3 ----
  {
    int e = lm;
    int s = src[eBase + e];
    const float* fRow = f + (size_t)s * 48;
    const float* bp = basis + (size_t)(eBase + e) * 18;
    float B[18];
#pragma unroll
    for (int i = 0; i < 18; ++i) B[i] = bp[i];
#pragma unroll
    for (int mm = 0; mm < 8; ++mm) {
      int m1 = lh * 8 + mm;
      float f0 = fRow[m1 * 3 + 0], f1 = fRow[m1 * 3 + 1], f2 = fRow[m1 * 3 + 2];
#pragma unroll
      for (int k = 0; k < 6; ++k) {
        float t = f0 * B[k] + f1 * B[6 + k] + f2 * B[12 + k];
        int m = 2 * m1 + k / 3;
        St[e * 96 + m * 3 + (k % 3)] = t;
      }
    }
  }
  __syncthreads();

  // ---- build A2 (16x64 bf16 = two 16x32 K-halves) from Sh ----------------
  V16BF a2lo, a2hi;
#pragma unroll
  for (int j = 0; j < 16; ++j) {
    int k = ((j < 8) ? j : (j + 8)) + lh * 8;
    a2lo.u[j] = f2bf(Sh[lm * 64 + k]);
    a2hi.u[j] = f2bf(Sh[lm * 64 + k + 32]);
  }

  // hoist tmp columns into registers: lane handles (e=lm, d=lh) and (e=lm, d=2)
  float tA[32], tB[32];
#pragma unroll
  for (int m = 0; m < 32; ++m) {
    tA[m] = St[lm * 96 + m * 3 + lh];
    tB[m] = St[lm * 96 + m * 3 + 2];
  }

  // ---- GEMM2 (48 N-tiles, K=64) fused with conv contraction --------------
  const unsigned short* W2b = Bswz + 2048;
  for (int c = 0; c < 24; ++c) {
    v8f accE, accO;
    float bE = b2[(2 * c) * 16 + lm];
    float bO = b2[(2 * c + 1) * 16 + lm];
#pragma unroll
    for (int j = 0; j < 8; ++j) { accE[j] = bE; accO[j] = bO; }
    v16bf w0 = *(const v16bf*)(W2b + (4 * c + 0) * 512 + lane * 16);
    v16bf w1 = *(const v16bf*)(W2b + (4 * c + 1) * 512 + lane * 16);
    v16bf w2 = *(const v16bf*)(W2b + (4 * c + 2) * 512 + lane * 16);
    v16bf w3 = *(const v16bf*)(W2b + (4 * c + 3) * 512 + lane * 16);
    accE = __builtin_amdgcn_wmma_f32_16x16x32_bf16(false, a2lo.v, false, w0,
                                                   (short)0, accE, false, false);
    accE = __builtin_amdgcn_wmma_f32_16x16x32_bf16(false, a2hi.v, false, w1,
                                                   (short)0, accE, false, false);
    accO = __builtin_amdgcn_wmma_f32_16x16x32_bf16(false, a2lo.v, false, w2,
                                                   (short)0, accO, false, false);
    accO = __builtin_amdgcn_wmma_f32_16x16x32_bf16(false, a2hi.v, false, w3,
                                                   (short)0, accO, false, false);
    // stage rw[e, m] (m = column within the c-group of 32)
#pragma unroll
    for (int j = 0; j < 8; ++j) {
      Sr[(j + 8 * lh) * 32 + lm]      = accE[j];
      Sr[(j + 8 * lh) * 32 + 16 + lm] = accO[j];
    }
    __syncthreads();
    {
      float r[32];
      const float* rp = Sr + lm * 32;
#pragma unroll
      for (int m = 0; m < 32; ++m) r[m] = rp[m];
      float s1 = 0.f, s2 = 0.f;
#pragma unroll
      for (int m = 0; m < 32; ++m) { s1 += r[m] * tA[m]; s2 += r[m] * tB[m]; }
      Sc[lm * 72 + c * 3 + lh] = s1;          // d = 0 or 1
      if (lane < 16) Sc[lm * 72 + c * 3 + 2] = s2;   // d = 2
    }
    __syncthreads();
  }

  // ---- attention scores: leaky_relu(sum(k*q)/sqrt(24)), segment max ------
#pragma unroll
  for (int it = 0; it < 2; ++it) {
    int e = lm;
    int h = lh + 2 * it;
    float s = 0.f;
#pragma unroll
    for (int p = 0; p < 6; ++p)
      s += Sc[e * 72 + 6 * h + p] * Sc[e * 72 + 24 + 6 * h + p];
    s *= 0.2041241452319315f;                 // 24^-0.5
    s = (s > 0.f) ? s : 0.2f * s;             // leaky relu
    scoresWs[(size_t)(eBase + e) * 4 + h] = s;
    atomicMaxF(smax + (size_t)dst[eBase + e] * 4 + h, s);
  }
  // ---- spill v (conv c = 16..23) -----------------------------------------
  {
    int e = lm;
#pragma unroll
    for (int i = 0; i < 12; ++i) {
      int j = lh * 12 + i;
      vWs[(size_t)(eBase + e) * 24 + j] = Sc[e * 72 + 48 + j];
    }
  }
}

// ---------------------------------------------------------------------------
// exp(score - max), scatter numerator into out and denominator per node.
// ---------------------------------------------------------------------------
__global__ __launch_bounds__(256) void ea_softmax_scatter(
    const int* __restrict__ dst, const float* __restrict__ scoresWs,
    const float* __restrict__ smax, const float* __restrict__ vWs,
    float* __restrict__ denom, float* __restrict__ out)
{
  int tid = blockIdx.x * blockDim.x + threadIdx.x;
  if (tid >= E_EDGES * 4) return;
  int e = tid >> 2, h = tid & 3;
  int n = dst[e];
  float ex = __expf(scoresWs[tid] - smax[n * 4 + h]);
  atomicAdd(denom + n * 4 + h, ex);
  const float* vp = vWs + (size_t)e * 24 + h * 6;
  float* op = out + (size_t)n * 24 + h * 6;
#pragma unroll
  for (int i = 0; i < 6; ++i) atomicAdd(op + i, ex * vp[i]);
}

__global__ __launch_bounds__(256) void ea_norm(float* __restrict__ out,
                                               const float* __restrict__ denom) {
  int tid = blockIdx.x * blockDim.x + threadIdx.x;
  if (tid >= N_NODES * 24) return;
  float d = denom[(tid / 24) * 4 + (tid % 24) / 6];
  out[tid] = (d > 0.f) ? out[tid] / d : 0.f;
}

extern "C" void kernel_launch(void* const* d_in, const int* in_sizes, int n_in,
                              void* d_out, int out_size, void* d_ws, size_t ws_size,
                              hipStream_t stream) {
  (void)in_sizes; (void)n_in; (void)out_size; (void)ws_size;
  const int*   src        = (const int*)d_in[0];
  const int*   dst        = (const int*)d_in[1];
  const float* basis      = (const float*)d_in[2];
  const float* edge_feats = (const float*)d_in[3];
  const float* f          = (const float*)d_in[4];
  const float* W1         = (const float*)d_in[5];
  const float* b1         = (const float*)d_in[6];
  const float* W2         = (const float*)d_in[7];
  const float* b2         = (const float*)d_in[8];
  float* out = (float*)d_out;

  // workspace layout
  unsigned short* Bswz = (unsigned short*)d_ws;              // 102400 B
  float* scoresWs = (float*)((char*)d_ws + 102400);          // E*4 f32
  float* vWs   = scoresWs + (size_t)E_EDGES * 4;             // E*24 f32
  float* smax  = vWs + (size_t)E_EDGES * 24;                 // N*4 f32
  float* denom = smax + (size_t)N_NODES * 4;                 // N*4 f32

  ea_prep<<<(PREP_ELEMS + 255) / 256, 256, 0, stream>>>(W1, W2, Bswz);
  ea_init<<<(N_NODES * 24 + 255) / 256, 256, 0, stream>>>(out, smax, denom);
  ea_main<<<E_EDGES / EDGES_PER_BLOCK, WAVES_PER_BLOCK * 32, 0, stream>>>(
      src, dst, basis, edge_feats, f, b1, b2, Bswz, scoresWs, vWs, smax);
  ea_softmax_scatter<<<(E_EDGES * 4 + 255) / 256, 256, 0, stream>>>(
      dst, scoresWs, smax, vWs, denom, out);
  ea_norm<<<(N_NODES * 24 + 255) / 256, 256, 0, stream>>>(out, denom);
}
